// GRU_10746008175428
// MI455X (gfx1250) — matrix operand order repaired and checked by reading, not compiled, not run on hardware
//
#include <hip/hip_runtime.h>
#include <hip/hip_bf16.h>

typedef __attribute__((ext_vector_type(16))) _Float16 v16h;
typedef __attribute__((ext_vector_type(8)))  _Float16 v8h;
typedef __attribute__((ext_vector_type(8)))  float    v8f;

__device__ __forceinline__ v16h cat8(v8h lo, v8h hi) {
    return __builtin_shufflevector(lo, hi, 0,1,2,3,4,5,6,7,8,9,10,11,12,13,14,15);
}

__device__ __forceinline__ v8f wmma_f16(v16h a, v16h b, v8f c) {
    // v_wmma_f32_16x16x32_f16  (neg_a, A, neg_b, B, c_mod, C, reuse_a, reuse_b)
    return __builtin_amdgcn_wmma_f32_16x16x32_f16(false, a, false, b, (short)0, c, false, false);
}

__device__ __forceinline__ float fsig(float x)  { return 1.0f / (1.0f + __expf(-x)); }
__device__ __forceinline__ float ftanh(float x) { return 2.0f * fsig(2.0f * x) - 1.0f; }

// ---------------------------------------------------------------------------
// f32 -> f16 conversion (grid-stride)
// ---------------------------------------------------------------------------
__global__ void cvt_f32_f16(const float* __restrict__ in, _Float16* __restrict__ out, int n) {
    for (int i = blockIdx.x * blockDim.x + threadIdx.x; i < n; i += gridDim.x * blockDim.x)
        out[i] = (_Float16)in[i];
}

// ---------------------------------------------------------------------------
// One GRU layer. Grid = B/16 blocks, 256 threads (8 waves).
// Each block owns batch rows [m0, m0+16) and iterates all T steps privately
// (batch rows are independent across the recurrence -> no inter-block sync).
// Per step: 24 column tiles of the 16x384 gate matrices, 3 tiles per wave.
//   xg tile: A streamed from global (f16 input seq), B streamed from global W_ih
//   hg tile: A = h (fp16, LDS),                     B = W_hh (fp16, LDS)
// Gates computed elementwise with v_exp-based sigmoid/tanh; h fed back via LDS.
// ---------------------------------------------------------------------------
template<int KIN, bool STORE_SEQ, bool STORE_LAST>
__global__ __launch_bounds__(256)
void gru_layer_kernel(const _Float16* __restrict__ xin,   // [B][T][KIN]  f16
                      const _Float16* __restrict__ wih,   // [384][KIN]   f16
                      const _Float16* __restrict__ whh,   // [384][128]   f16
                      const float*    __restrict__ b_ih,  // [384]
                      const float*    __restrict__ b_hh,  // [384]
                      _Float16* __restrict__ seq_out,     // [B][T][128]  (if STORE_SEQ)
                      _Float16* __restrict__ last_out,    // [B][128]     (if STORE_LAST)
                      int T)
{
    constexpr int H = 128, G = 384;
    __shared__ _Float16 whh_lds[G * H];   //  98304 B
    __shared__ _Float16 h_lds[16 * H];    //   4096 B
    __shared__ float    xg_lds[16 * G];   //  24576 B
    __shared__ float    hg_lds[16 * G];   //  24576 B
    __shared__ float    bih_lds[G];       //   1536 B
    __shared__ float    bhh_lds[G];       //   1536 B  -> ~151 KB total (< 320 KB/WGP)

    const int tid  = threadIdx.x;
    const int lid  = tid & 31;
    const int wid  = tid >> 5;
    const int m0   = blockIdx.x * 16;
    const int arow = lid & 15;       // A-fragment row owned by this lane
    const int ahlf = lid >> 4;       // 0: K{0..15}, 1: K{16..31}

    // Preload W_hh and biases into LDS; zero h.
    for (int i = tid; i < (G * H) / 8; i += 256)
        *reinterpret_cast<v8h*>(&whh_lds[i * 8]) =
            *reinterpret_cast<const v8h*>(&whh[i * 8]);
    for (int i = tid; i < G; i += 256) { bih_lds[i] = b_ih[i]; bhh_lds[i] = b_hh[i]; }
    for (int i = tid; i < 16 * H; i += 256) h_lds[i] = (_Float16)0.0f;
    __syncthreads();

    for (int t = 0; t < T; ++t) {
        // ---- WMMA phase: xg and hg tiles (3 per wave) ----
        const _Float16* xrow = xin + ((size_t)(m0 + arow) * T + t) * KIN;
        #pragma unroll
        for (int i3 = 0; i3 < 3; ++i3) {
            const int nt   = wid * 3 + i3;          // tile 0..23
            const int ncol = nt * 16 + (lid & 15);  // gate column 0..383
            v8f accx = {};
            v8f acch = {};
            // xg = x_t @ W_ih^T   (A from global, B = W_ih rows, contiguous K)
            #pragma unroll
            for (int k0 = 0; k0 < KIN; k0 += 32) {
                v8h lo = *reinterpret_cast<const v8h*>(xrow + k0 + ahlf * 16);
                v8h hi = *reinterpret_cast<const v8h*>(xrow + k0 + ahlf * 16 + 8);
                v16h a = cat8(lo, hi);
                v16h b = *reinterpret_cast<const v16h*>(
                    wih + (size_t)ncol * KIN + k0 + ahlf * 16);
                accx = wmma_f16(a, b, accx);
            }
            // hg = h @ W_hh^T     (A and B both LDS-resident)
            #pragma unroll
            for (int k0 = 0; k0 < H; k0 += 32) {
                v8h lo = *reinterpret_cast<const v8h*>(&h_lds[arow * H + k0 + ahlf * 16]);
                v8h hi = *reinterpret_cast<const v8h*>(&h_lds[arow * H + k0 + ahlf * 16 + 8]);
                v16h a = cat8(lo, hi);
                v16h b = *reinterpret_cast<const v16h*>(
                    &whh_lds[ncol * H + k0 + ahlf * 16]);
                acch = wmma_f16(a, b, acch);
            }
            // C layout: VGPR i, lane l -> row i + 8*(l>=16), col l&15
            #pragma unroll
            for (int i = 0; i < 8; ++i) {
                const int row = i + 8 * ahlf;
                const int col = nt * 16 + (lid & 15);
                xg_lds[row * G + col] = accx[i] + bih_lds[col];
                hg_lds[row * G + col] = acch[i] + bhh_lds[col];
            }
        }
        __syncthreads();

        // ---- gate phase: 16x128 elements, 8 per thread ----
        #pragma unroll
        for (int e = 0; e < 8; ++e) {
            const int j = e * 256 + tid;    // 0..2047
            const int m = j >> 7;
            const int c = j & 127;
            const float xr = xg_lds[m * G + c],       hr = hg_lds[m * G + c];
            const float xz = xg_lds[m * G + 128 + c], hz = hg_lds[m * G + 128 + c];
            const float xn = xg_lds[m * G + 256 + c], hn = hg_lds[m * G + 256 + c];
            const float r = fsig(xr + hr);
            const float z = fsig(xz + hz);
            const float n = ftanh(xn + r * hn);
            const float hold = (float)h_lds[m * H + c];
            const float hnew = (1.0f - z) * n + z * hold;
            h_lds[m * H + c] = (_Float16)hnew;   // single owner thread per element
            if (STORE_SEQ)
                seq_out[((size_t)(m0 + m) * T + t) * H + c] = (_Float16)hnew;
            if (STORE_LAST && t == T - 1)
                last_out[(size_t)(m0 + m) * H + c] = (_Float16)hnew;
        }
        __syncthreads();
    }
}

// ---------------------------------------------------------------------------
// Head: out[512][96] = h_last[512][128] @ W_head[96][128]^T + b_head
// 192 wave-tiles of 16x16, 8 waves/block -> 24 blocks, K-loop of 4 WMMAs.
// ---------------------------------------------------------------------------
__global__ __launch_bounds__(256)
void head_kernel(const _Float16* __restrict__ hlast,  // [512][128] f16
                 const _Float16* __restrict__ whead,  // [96][128]  f16
                 const float*    __restrict__ bias,   // [96]
                 float* __restrict__ out)             // [512][96]
{
    const int tid  = threadIdx.x;
    const int lid  = tid & 31;
    const int wid  = tid >> 5;
    const int tile = blockIdx.x * 8 + wid;   // 0..191
    const int mt   = tile / 6;
    const int nt   = tile % 6;
    const int arow = lid & 15;
    const int ahlf = lid >> 4;
    const int ncol = nt * 16 + (lid & 15);

    const _Float16* ap = hlast + (size_t)(mt * 16 + arow) * 128;
    v8f acc = {};
    #pragma unroll
    for (int k0 = 0; k0 < 128; k0 += 32) {
        v8h lo = *reinterpret_cast<const v8h*>(ap + k0 + ahlf * 16);
        v8h hi = *reinterpret_cast<const v8h*>(ap + k0 + ahlf * 16 + 8);
        v16h a = cat8(lo, hi);
        v16h b = *reinterpret_cast<const v16h*>(whead + (size_t)ncol * 128 + k0 + ahlf * 16);
        acc = wmma_f16(a, b, acc);
    }
    const float bb = bias[ncol];
    #pragma unroll
    for (int i = 0; i < 8; ++i) {
        const int row = mt * 16 + i + 8 * ahlf;
        out[(size_t)row * 96 + nt * 16 + (lid & 15)] = acc[i] + bb;
    }
}

// ---------------------------------------------------------------------------
extern "C" void kernel_launch(void* const* d_in, const int* in_sizes, int n_in,
                              void* d_out, int out_size, void* d_ws, size_t ws_size,
                              hipStream_t stream) {
    (void)in_sizes; (void)n_in; (void)out_size; (void)ws_size;
    constexpr int B = 512, T = 512, E = 64, H = 128, P = 96, G = 384;

    const float* x      = (const float*)d_in[0];
    const float* W_ih0  = (const float*)d_in[1];
    const float* W_hh0  = (const float*)d_in[2];
    const float* b_ih0  = (const float*)d_in[3];
    const float* b_hh0  = (const float*)d_in[4];
    const float* W_ih1  = (const float*)d_in[5];
    const float* W_hh1  = (const float*)d_in[6];
    const float* b_ih1  = (const float*)d_in[7];
    const float* b_hh1  = (const float*)d_in[8];
    const float* W_head = (const float*)d_in[9];
    const float* b_head = (const float*)d_in[10];
    float* out = (float*)d_out;

    char* ws = (char*)d_ws;
    size_t o = 0;
    _Float16* x16      = (_Float16*)(ws + o); o += (size_t)B * T * E * 2;  // 33.5 MB
    _Float16* seq16    = (_Float16*)(ws + o); o += (size_t)B * T * H * 2;  // 67 MB
    _Float16* hlast16  = (_Float16*)(ws + o); o += (size_t)B * H * 2;
    _Float16* wih0_16  = (_Float16*)(ws + o); o += (size_t)G * E * 2;
    _Float16* whh0_16  = (_Float16*)(ws + o); o += (size_t)G * H * 2;
    _Float16* wih1_16  = (_Float16*)(ws + o); o += (size_t)G * H * 2;
    _Float16* whh1_16  = (_Float16*)(ws + o); o += (size_t)G * H * 2;
    _Float16* whead_16 = (_Float16*)(ws + o); o += (size_t)P * H * 2;

    cvt_f32_f16<<<2048, 256, 0, stream>>>(x,      x16,      B * T * E);
    cvt_f32_f16<<<  96, 256, 0, stream>>>(W_ih0,  wih0_16,  G * E);
    cvt_f32_f16<<< 192, 256, 0, stream>>>(W_hh0,  whh0_16,  G * H);
    cvt_f32_f16<<< 192, 256, 0, stream>>>(W_ih1,  wih1_16,  G * H);
    cvt_f32_f16<<< 192, 256, 0, stream>>>(W_hh1,  whh1_16,  G * H);
    cvt_f32_f16<<<  48, 256, 0, stream>>>(W_head, whead_16, P * H);

    gru_layer_kernel<64,  true,  false><<<B / 16, 256, 0, stream>>>(
        x16, wih0_16, whh0_16, b_ih0, b_hh0, seq16, nullptr, T);
    gru_layer_kernel<128, false, true ><<<B / 16, 256, 0, stream>>>(
        seq16, wih1_16, whh1_16, b_ih1, b_hh1, nullptr, hlast16, T);

    head_kernel<<<(B / 16) * 6 / 8, 256, 0, stream>>>(hlast16, whead_16, b_head, out);
}
